// RCModel_41463614275898
// MI455X (gfx1250) — matrix operand order repaired and checked by reading, not compiled, not run on hardware
//
#include <hip/hip_runtime.h>

typedef __attribute__((ext_vector_type(16))) _Float16 v16h;
typedef __attribute__((ext_vector_type(8)))  _Float16 v8h;
typedef __attribute__((ext_vector_type(8)))  float    v8f;

#define NS   256      // state dim
#define NH   64       // hidden dim
#define NRM  254      // rooms
#define MCOL 255      // B columns
#define MU_C   23.359f
#define ISTD_C (1.0f/1.41f)
#define DSCALE 256.0f
#define IDSCALE (1.0f/256.0f)

union AU { _Float16 h[16]; v16h v; };
union BU { unsigned int u[8]; v16h v; };

// ---------------- precompute: Z = X*Y (256x256 fp32) ----------------
__global__ void rc_mm(const float* __restrict__ X, const float* __restrict__ Y,
                      float* __restrict__ Z) {
    __shared__ float xr[NS];
    int i = blockIdx.x, j = threadIdx.x;
    xr[j] = X[i * NS + j];
    __syncthreads();
    float acc = 0.f;
    for (int k = 0; k < NS; ++k) acc = fmaf(xr[k], Y[k * NS + j], acc);
    Z[i * NS + j] = acc;
}

// ---------------- precompute: D = dtA + dt^2/2 A2 + dt^3/6 A3 + dt^4/24 A4 ----
__global__ void rc_dmat(const float* __restrict__ A, const float* __restrict__ A2,
                        const float* __restrict__ A3, const float* __restrict__ A4,
                        const float* __restrict__ t_eval, float* __restrict__ D) {
    int idx = blockIdx.x * NS + threadIdx.x;
    float dt = t_eval[1] - t_eval[0];
    float c2 = 0.5f * dt * dt;
    float c3 = dt * dt * dt * (1.0f / 6.0f);
    float c4 = dt * dt * dt * dt * (1.0f / 24.0f);
    D[idx] = dt * A[idx] + c2 * A2[idx] + c3 * A3[idx] + c4 * A4[idx];
}

// ---------------- precompute: forcing vectors p1,p2,p4,v --------------------
__global__ void rc_vec(const float* __restrict__ A, const float* __restrict__ A2,
                       const float* __restrict__ A3, const float* __restrict__ Bm,
                       const float* __restrict__ cooling, const float* __restrict__ t_eval,
                       float* __restrict__ p1, float* __restrict__ p2,
                       float* __restrict__ p4, float* __restrict__ vout) {
    __shared__ float sheat[NS], sbq[NS], sb0[NS];
    int i = threadIdx.x;
    sheat[i] = (i < NRM) ? (500.0f / (1.0f + __expf(-cooling[i]))) : 0.0f;
    __syncthreads();
    float bq = 0.f;
    for (int r = 0; r < NRM; ++r) bq = fmaf(Bm[i * MCOL + 1 + r], sheat[r], bq);
    float b0 = Bm[i * MCOL];
    sbq[i] = bq; sb0[i] = b0;
    __syncthreads();
    float u1 = 0, u2 = 0, u3 = 0, x1 = 0, x2 = 0, x3 = 0;
    for (int j = 0; j < NS; ++j) {
        float a = A[i * NS + j], a2 = A2[i * NS + j], a3 = A3[i * NS + j];
        float qj = sbq[j], bj = sb0[j];
        u1 = fmaf(a, qj, u1); u2 = fmaf(a2, qj, u2); u3 = fmaf(a3, qj, u3);
        x1 = fmaf(a, bj, x1); x2 = fmaf(a2, bj, x2); x3 = fmaf(a3, bj, x3);
    }
    float dt = t_eval[1] - t_eval[0];
    float c = dt * (1.0f / 6.0f);
    float dt2 = dt * dt, dt3 = dt2 * dt;
    p4[i]   = c * b0;
    p2[i]   = c * (4.0f * b0 + 2.0f * dt * x1 + 0.5f * dt2 * x2);
    p1[i]   = c * (b0 + dt * x1 + 0.5f * dt2 * x2 + 0.25f * dt3 * x3);
    vout[i] = -c * (6.0f * bq + 3.0f * dt * u1 + dt2 * u2 + 0.25f * dt3 * u3);
}

// ------- precompute: pre[k][h] = b1[h] + w1[h,256:260] . timefeatures(t_k) ----
__global__ void rc_pre(const float* __restrict__ w1, const float* __restrict__ b1,
                       const float* __restrict__ t_eval, float* __restrict__ pre, int T) {
    int gid = blockIdx.x * 256 + threadIdx.x;
    if (gid >= T * NH) return;
    int k = gid >> 6, h = gid & 63;
    float dt = t_eval[1] - t_eval[0];
    float ts = (float)k * dt;
    const float TWOPI = 6.28318530717958647692f;
    float wd = TWOPI / 86400.0f, ww = TWOPI / 604800.0f;
    float s0 = sinf(ts * wd), c0 = cosf(ts * wd);
    float s1 = sinf(ts * ww), c1 = cosf(ts * ww);
    const float* wr = &w1[h * (NS + 4) + NS];
    pre[gid] = b1[h] + wr[0] * s0 + wr[1] * c0 + wr[2] * s1 + wr[3] * c1;
}

// ---------------- main persistent sequential integrator ----------------------
__global__ __launch_bounds__(256)
void rc_main(const float* __restrict__ t_eval, const float* __restrict__ w1,
             const float* __restrict__ w2, const float* __restrict__ tts,
             const float* __restrict__ tvals, const float* __restrict__ Dm,
             const float* __restrict__ p1v, const float* __restrict__ p2v,
             const float* __restrict__ p4v, const float* __restrict__ vv,
             const float* __restrict__ pre, float* __restrict__ out,
             int T, int ntout) {
    __shared__ _Float16 W1xh[NH * 264];   // policy weights, fp16, padded stride
    __shared__ _Float16 xhL[NS];          // x as fp16 (WMMA B operand source)
    __shared__ _Float16 xnhL[NS];         // normalized x as fp16 (policy input)
    __shared__ float yvL[NS], partL[NS], hidL[NH], toutV[2048];
    __shared__ float aS;

    const int tid  = threadIdx.x;
    const int lane = tid & 31;
    const int wv   = tid >> 5;            // wave 0..7, owns row-tiles 2wv, 2wv+1
    const int hf   = lane >> 4;           // lane half
    const int r16  = lane & 15;

    float dt  = t_eval[1] - t_eval[0];
    float t0  = t_eval[0];
    float ts0 = tts[0];
    float dTs = tts[1] - tts[0];
    float invdTs = 1.0f / dTs;            // one divide total; loop uses multiplies

    // ---- one-time init of LDS ----
    for (int idx = tid; idx < NH * NS; idx += 256) {
        int h = idx >> 8, j = idx & 255;
        W1xh[h * 264 + j] = (_Float16)w1[h * (NS + 4) + j];
    }
    for (int idx = tid; idx < ntout; idx += 256) toutV[idx] = tvals[idx];

    // ---- gather loop-invariant A-operands (D tiles, fp16, scaled) into VGPRs ----
    // 16-bit A 16x32 layout: lane (hf,r16) holds row M=r16; vgpr v: K = 2v + (v>=4?8:0) + 8*hf
    v16h areg[16];
#pragma unroll
    for (int ti = 0; ti < 2; ++ti) {
        int row = (2 * wv + ti) * 16 + r16;
#pragma unroll
        for (int c = 0; c < 8; ++c) {
            AU au;
#pragma unroll
            for (int v8i = 0; v8i < 8; ++v8i) {
                int K = 32 * c + 2 * v8i + ((v8i >= 4) ? 8 : 0) + 8 * hf;
                au.h[2 * v8i]     = (_Float16)(DSCALE * Dm[row * NS + K]);
                au.h[2 * v8i + 1] = (_Float16)(DSCALE * Dm[row * NS + K + 1]);
            }
            areg[ti * 8 + c] = au.v;
        }
    }

    // per-thread loop-invariant vectors
    float vr  = vv[tid], p1r = p1v[tid], p2r = p2v[tid], p4r = p4v[tid];
    float w2a = 0.f, w2b = 0.f;
    if (tid < 32) { w2a = w2[tid]; w2b = w2[tid + 32]; }
    float preNext = (tid < NH) ? pre[tid] : 0.f;

    float xi = 26.0f;
    xhL[tid]  = (_Float16)xi;
    xnhL[tid] = (_Float16)((xi - MU_C) * ISTD_C);
    __syncthreads();

    // uniform-grid interp into shared tout table (no divides in the loop)
    auto itp = [&](float t) -> float {
        float pos = (t - ts0) * invdTs;
        int idx = (int)floorf(pos);
        idx = idx < 0 ? 0 : (idx > ntout - 2 ? ntout - 2 : idx);
        float w = (t - (ts0 + (float)idx * dTs)) * invdTs;
        return fmaf(w, toutV[idx + 1] - toutV[idx], toutV[idx]);
    };
    float To1 = itp(t0), To2 = itp(t0 + 0.5f * dt), To4 = itp(t0 + dt);

    const int h = tid & 63, q = tid >> 6;

    for (int k = 0; k < T; ++k) {
        // emit state AT t (reference emits pre-step state)
        out[(size_t)k * NS + tid] = xi;

        // ---- y = (256*D) @ x via fp16 WMMA, x replicated across all 16 B columns ----
        v8f acc0 = {}, acc1 = {};
        const unsigned int* xhu = (const unsigned int*)xhL;
#pragma unroll
        for (int c = 0; c < 8; ++c) {
            BU bu;
#pragma unroll
            for (int v8i = 0; v8i < 8; ++v8i)
                bu.u[v8i] = xhu[16 * c + v8i + ((v8i >= 4) ? 4 : 0) + 4 * hf];   // pairs (K, K+1), K = 32c+2v+(v>=4?8:0)+8hf
            acc0 = __builtin_amdgcn_wmma_f32_16x16x32_f16(
                false, areg[c],     false, bu.v, (short)0, acc0, false, false);
            acc1 = __builtin_amdgcn_wmma_f32_16x16x32_f16(
                false, areg[8 + c], false, bu.v, (short)0, acc1, false, false);
        }

        // ---- policy partial dot (fp32 accumulate over fp16 LDS); 4 independent
        // accumulators to break the 64-deep FMA dependency chain seen in r1 asm
        {
            const v8h* wr = (const v8h*)&W1xh[h * 264 + q * 64];
            const v8h* xr = (const v8h*)&xnhL[q * 64];
            float s0 = 0.f, s1 = 0.f, s2 = 0.f, s3 = 0.f;
#pragma unroll
            for (int j8 = 0; j8 < 8; ++j8) {
                v8h av = wr[j8], bv = xr[j8];
                s0 = fmaf((float)av[0], (float)bv[0], s0);
                s1 = fmaf((float)av[1], (float)bv[1], s1);
                s2 = fmaf((float)av[2], (float)bv[2], s2);
                s3 = fmaf((float)av[3], (float)bv[3], s3);
                s0 = fmaf((float)av[4], (float)bv[4], s0);
                s1 = fmaf((float)av[5], (float)bv[5], s1);
                s2 = fmaf((float)av[6], (float)bv[6], s2);
                s3 = fmaf((float)av[7], (float)bv[7], s3);
            }
            partL[tid] = (s0 + s1) + (s2 + s3);
        }

        // ---- write y tiles (columns replicated -> lanes 0 and 16 hold everything)
        if ((lane & 15) == 0) {
            int off = hf * 8, s0t = 2 * wv * 16, s1t = (2 * wv + 1) * 16;
#pragma unroll
            for (int j = 0; j < 8; ++j) {
                yvL[s0t + off + j] = acc0[j];
                yvL[s1t + off + j] = acc1[j];
            }
        }
        __syncthreads();

        if (tid < NH) {
            float hs = (partL[tid] + partL[64 + tid]) + (partL[128 + tid] + partL[192 + tid])
                     + preNext;
            hidL[tid] = tanhf(hs);
        }
        __syncthreads();

        if (tid < 32) {
            float s = fmaf(w2a, hidL[tid], w2b * hidL[tid + 32]);
            s += __shfl_xor(s, 16); s += __shfl_xor(s, 8);
            s += __shfl_xor(s, 4);  s += __shfl_xor(s, 2); s += __shfl_xor(s, 1);
            if (tid == 0) aS = 1.0f / (1.0f + __expf(-s));
        }
        __syncthreads();

        // ---- exact RK4 update: x+ = x + D x + a*v + To1*p1 + To2*p2 + To4*p4
        float a = aS;
        float g = To1 * p1r + To2 * p2r + To4 * p4r;
        xi = xi + IDSCALE * yvL[tid] + a * vr + g;
        xhL[tid]  = (_Float16)xi;
        xnhL[tid] = (_Float16)((xi - MU_C) * ISTD_C);

        // prefetch next-step MLP bias; compute next-step Tout triple (off critical path)
        if (tid < NH && (k + 1) < T) preNext = pre[(size_t)(k + 1) * NH + tid];
        float tn = t0 + (float)(k + 1) * dt;
        To1 = itp(tn); To2 = itp(tn + 0.5f * dt); To4 = itp(tn + dt);
        __syncthreads();
    }
}

extern "C" void kernel_launch(void* const* d_in, const int* in_sizes, int n_in,
                              void* d_out, int out_size, void* d_ws, size_t ws_size,
                              hipStream_t stream) {
    const float* t_eval  = (const float*)d_in[0];
    const float* A       = (const float*)d_in[1];
    const float* Bm      = (const float*)d_in[2];
    const float* cooling = (const float*)d_in[3];
    const float* w1      = (const float*)d_in[4];
    const float* b1      = (const float*)d_in[5];
    const float* w2      = (const float*)d_in[6];
    const float* tts     = (const float*)d_in[7];
    const float* tvals   = (const float*)d_in[8];
    int T     = in_sizes[0];
    int ntout = in_sizes[7];

    float* ws  = (float*)d_ws;
    float* A2  = ws;
    float* A3  = ws + 65536;
    float* A4  = ws + 131072;
    float* Dm  = ws + 196608;
    float* p1  = ws + 262144;
    float* p2  = p1 + NS;
    float* p4  = p2 + NS;
    float* vv  = p4 + NS;
    float* pre = vv + NS;   // T*64 floats

    rc_mm  <<<NS, NS, 0, stream>>>(A,  A, A2);
    rc_mm  <<<NS, NS, 0, stream>>>(A2, A, A3);
    rc_mm  <<<NS, NS, 0, stream>>>(A3, A, A4);
    rc_dmat<<<NS, NS, 0, stream>>>(A, A2, A3, A4, t_eval, Dm);
    rc_vec <<<1,  NS, 0, stream>>>(A, A2, A3, Bm, cooling, t_eval, p1, p2, p4, vv);
    rc_pre <<<(T * NH + 255) / 256, 256, 0, stream>>>(w1, b1, t_eval, pre, T);
    rc_main<<<1,  NS, 0, stream>>>(t_eval, w1, w2, tts, tvals, Dm,
                                   p1, p2, p4, vv, pre, (float*)d_out, T, ntout);
}
